// decoder_mix_20358144983647
// MI455X (gfx1250) — matrix operand order repaired and checked by reading, not compiled, not run on hardware
//
#include <hip/hip_runtime.h>

typedef unsigned short u16;
typedef __attribute__((ext_vector_type(16))) __bf16 bf16x16;
typedef __attribute__((ext_vector_type(8)))  float  floatx8;
typedef __attribute__((ext_vector_type(4)))  unsigned u32x4;
typedef __attribute__((ext_vector_type(4)))  int      i32x4;

#define GAS __attribute__((address_space(1)))
#define LAS __attribute__((address_space(3)))

#if defined(__AMDGCN__) && __has_builtin(__builtin_amdgcn_global_load_async_to_lds_b128) && __has_builtin(__builtin_amdgcn_s_wait_asynccnt)
#define HAVE_ASYNC_LDS 1
#else
#define HAVE_ASYNC_LDS 0
#endif

static __device__ __forceinline__ u16 f2bf(float f) {
    unsigned u = __builtin_bit_cast(unsigned, f);
    unsigned r = u + 0x7FFFu + ((u >> 16) & 1u);   // round-to-nearest-even
    return (u16)(r >> 16);
}

union FragCvt { struct { u32x4 lo, hi; } q; bf16x16 v; };
static __device__ __forceinline__ bf16x16 make_frag(u32x4 lo, u32x4 hi) {
    FragCvt u; u.q.lo = lo; u.q.hi = hi; return u.v;
}

// 16-byte global -> LDS copy: async on gfx1250 (no VGPR round trip, ASYNCcnt),
// synchronous fallback otherwise. Builtin params are (global i32x4*, local i32x4*,
// imm offset, imm cpol) per the compiler's diagnostic.
static __device__ __forceinline__ void copy16_g2l(u16* lds_dst, const u16* gsrc) {
#if HAVE_ASYNC_LDS
    __builtin_amdgcn_global_load_async_to_lds_b128(
        (GAS i32x4*)gsrc, (LAS i32x4*)lds_dst, 0, 0);
#else
    *(u32x4*)lds_dst = *(const u32x4*)gsrc;
#endif
}
static __device__ __forceinline__ void wait_async_copies() {
#if HAVE_ASYNC_LDS
    __builtin_amdgcn_s_wait_asynccnt(0);
#endif
}

// ---------------------------------------------------------------------------
// Elementwise f32 -> bf16
// ---------------------------------------------------------------------------
__global__ void cvt_bf16_kernel(const float* __restrict__ in, u16* __restrict__ out,
                                long long n) {
    long long i = (long long)blockIdx.x * blockDim.x + threadIdx.x;
    if (i < n) out[i] = f2bf(in[i]);
}

// f32 (batch, K, N) -> bf16 (batch, N, K)  (one-time weight transpose so GEMM
// B-tiles are contiguous row-major copies; no in-loop LDS scatter).
__global__ void cvt_bf16_T_kernel(const float* __restrict__ in, u16* __restrict__ out,
                                  int K, int N, long long total) {
    long long i = (long long)blockIdx.x * blockDim.x + threadIdx.x;
    if (i < total) {
        long long kn = (long long)K * N;
        long long b  = i / kn;
        long long r  = i - b * kn;
        int n = (int)(r / K);
        int k = (int)(r - (long long)n * K);
        out[i] = f2bf(in[b * kn + (long long)k * N + n]);
    }
}

// Extract z = s[..., 16:80] and convert to bf16. s rows are 80 f32 wide.
__global__ void pack_z_kernel(const float* __restrict__ s, u16* __restrict__ z,
                              long long n /* rows*64 */) {
    long long i = (long long)blockIdx.x * blockDim.x + threadIdx.x;
    if (i < n) {
        long long row = i >> 6;
        int k = (int)(i & 63);
        z[i] = f2bf(s[row * 80 + 16 + k]);
    }
}

// ---------------------------------------------------------------------------
// Batched bf16 GEMM + bias + activation (f32 accumulate, WMMA 16x16x32).
//   out[b,row,col] = act( sum_k A[b,row,k] * WT[b,col,k] + bias[b,col] )
// WT is the K-major (transposed) weight. 256 threads = 8 waves; block tile
// BM x BN; each wave computes 32x64 (2x4 WMMA accumulators). Double-buffered
// LDS staging via async global->LDS b128 copies.
// K % 32 == 0; rows % BM == 0; N % BN == 0. ACT: 0 none, 1 relu, 2 sigmoid.
// ---------------------------------------------------------------------------
template<int BM, int BN, int ACT, bool OUTBF16>
__global__ __launch_bounds__(256) void gemm_bf16_wmma(
    const u16* __restrict__ A, const u16* __restrict__ WT,
    const float* __restrict__ bias, void* __restrict__ outp,
    int K, int N,
    long long aBatch, long long wBatch, long long biasBatch, long long outBatch)
{
    constexpr int LD    = 40;          // LDS row stride in u16 (80B, 16B aligned)
    constexpr int ATILE = BM * LD;
    constexpr int BTILE = BN * LD;
    __shared__ u16 lds[2 * (ATILE + BTILE)];

    const int tid  = threadIdx.x;
    const int wave = tid >> 5;
    const int lane = tid & 31;
    constexpr int WGN = BN / 64;       // waves along N
    const int wm0 = (wave / WGN) * 32;
    const int wn0 = (wave % WGN) * 64;

    const int       batch   = blockIdx.z;
    const long long rowBase = (long long)blockIdx.y * BM;
    const int       colBase = blockIdx.x * BN;

    const u16*   Ab    = A  + batch * aBatch + rowBase * K;
    const u16*   Wb    = WT + batch * wBatch + (long long)colBase * K;
    const float* biasb = bias + batch * biasBatch;

    const int hl  = lane >> 4;
    const int l16 = lane & 15;

    floatx8 acc[2][4] = {};

    // Stage one K-chunk (32 wide) of A (BM x 32) and WT (BN x 32) into buffer.
    auto stage = [&](int kc, int buf) {
        u16* Al = lds + buf * (ATILE + BTILE);
        u16* Bl = Al + ATILE;
        #pragma unroll
        for (int i = 0; i < BM / 64; ++i) {        // BM*4 b128 units / 256 thr
            int u = tid + i * 256;
            int row = u >> 2, cq = (u & 3) * 8;
            copy16_g2l(&Al[row * LD + cq], Ab + (long long)row * K + kc + cq);
        }
        #pragma unroll
        for (int i = 0; i < BN / 64; ++i) {
            int u = tid + i * 256;
            int row = u >> 2, cq = (u & 3) * 8;
            copy16_g2l(&Bl[row * LD + cq], Wb + (long long)row * K + kc + cq);
        }
    };

    const int nk = K >> 5;
    stage(0, 0);
    wait_async_copies();
    __syncthreads();

    for (int t = 0; t < nk; ++t) {
        const int cur = t & 1;
        if (t + 1 < nk) stage((t + 1) * 32, cur ^ 1);   // prefetch next chunk

        const u16* Al = lds + cur * (ATILE + BTILE);
        const u16* Bl = Al + ATILE;

        bf16x16 af[2], bfr[4];
        #pragma unroll
        for (int i = 0; i < 2; ++i) {
            // A 16x32: lanes 0-15 hold K 0-15; lanes 16-31 K 16-31
            const u16* p = &Al[(wm0 + i * 16 + l16) * LD + 16 * hl];
            af[i] = make_frag(*(const u32x4*)p, *(const u32x4*)(p + 8));
        }
        #pragma unroll
        for (int j = 0; j < 4; ++j) {
            // B 32x16: element e of lane L is B[k = e + 16*(L/16)][n = L%16]
            const u16* p = &Bl[(wn0 + j * 16 + l16) * LD + 16 * hl];
            bfr[j] = make_frag(*(const u32x4*)p, *(const u32x4*)(p + 8));
        }

        #pragma unroll
        for (int i = 0; i < 2; ++i)
            #pragma unroll
            for (int j = 0; j < 4; ++j)
                acc[i][j] = __builtin_amdgcn_wmma_f32_16x16x32_bf16(
                    false, af[i], false, bfr[j], (short)0, acc[i][j], false, false);

        wait_async_copies();   // own wave's fills of the next buffer done
        __syncthreads();       // all waves done reading cur + filling next
    }

    // ---- epilogue: bias + activation + store
    // C/D layout: lane l16 = column, VGPR r = row (r + 8*hl) in 16x16 tile.
    #pragma unroll
    for (int i = 0; i < 2; ++i) {
        #pragma unroll
        for (int j = 0; j < 4; ++j) {
            int   col = colBase + wn0 + j * 16 + l16;
            float bv  = biasb[col];
            long long r0 = rowBase + wm0 + i * 16 + 8 * hl;
            #pragma unroll
            for (int r = 0; r < 8; ++r) {
                float x = acc[i][j][r] + bv;
                if (ACT == 1)      x = fmaxf(x, 0.0f);
                else if (ACT == 2) x = 1.0f / (1.0f + __expf(-x));
                long long idx = batch * outBatch + (r0 + r) * (long long)N + col;
                if (OUTBF16) ((u16*)outp)[idx]   = f2bf(x);
                else         ((float*)outp)[idx] = x;
            }
        }
    }
}

// ---------------------------------------------------------------------------
// Launcher
// ---------------------------------------------------------------------------
extern "C" void kernel_launch(void* const* d_in, const int* in_sizes, int n_in,
                              void* d_out, int out_size, void* d_ws, size_t ws_size,
                              hipStream_t stream) {
    (void)in_sizes; (void)n_in; (void)out_size; (void)ws_size;

    const float* s   = (const float*)d_in[0];  // (16, 2048, 80)
    const float* Wzz = (const float*)d_in[1];  // (16, 64, 64)
    const float* bzz = (const float*)d_in[2];  // (16, 64)
    const float* Wzh = (const float*)d_in[3];  // (16, 64, 512)
    const float* bzh = (const float*)d_in[4];  // (16, 512)
    const float* Whd = (const float*)d_in[5];  // (512, 512)
    const float* bhd = (const float*)d_in[6];  // (512,)
    const float* Whx = (const float*)d_in[7];  // (512, 3072)
    const float* bhx = (const float*)d_in[8];  // (3072,)
    float* out = (float*)d_out;                // (16, 2048, 3072)

    const long long ROWS = 16LL * 2048;        // 32768
    // workspace carve-up (bf16 elements; all offsets 16B-aligned)
    u16* w    = (u16*)d_ws;
    u16* zb   = w;  w += ROWS * 64;            // z in bf16
    u16* zz   = w;  w += ROWS * 64;            // stage-1 out
    u16* h1   = w;  w += ROWS * 512;           // stage-2 out
    u16* h2   = w;  w += ROWS * 512;           // stage-3 out
    u16* wzzT = w;  w += 16LL * 64 * 64;       // (16, 64, 64)   n-major
    u16* wzhT = w;  w += 16LL * 512 * 64;      // (16, 512, 64)  n-major
    u16* whdT = w;  w += 512LL * 512;          // (512, 512)     n-major
    u16* whxT = w;  w += 3072LL * 512;         // (3072, 512)    n-major

    {
        long long n = ROWS * 64;
        pack_z_kernel<<<dim3((unsigned)((n + 255) / 256)), 256, 0, stream>>>(s, zb, n);
    }
    auto cvtT = [&](const float* in, u16* o, int K, int N, long long total) {
        cvt_bf16_T_kernel<<<dim3((unsigned)((total + 255) / 256)), 256, 0, stream>>>(
            in, o, K, N, total);
    };
    cvtT(Wzz, wzzT, 64,  64,   16LL * 64 * 64);
    cvtT(Wzh, wzhT, 64,  512,  16LL * 64 * 512);
    cvtT(Whd, whdT, 512, 512,  512LL * 512);
    cvtT(Whx, whxT, 512, 3072, 512LL * 3072);

    // ---- stage 1: zz = z @ Wzz + bzz  (per-m, 2048x64x64)
    gemm_bf16_wmma<256, 64, 0, true><<<dim3(1, 2048 / 256, 16), 256, 0, stream>>>(
        zb, wzzT, bzz, zz, 64, 64,
        2048LL * 64, 64LL * 64, 64LL, 2048LL * 64);

    // ---- stage 2: h1 = relu(zz @ Wzh + bzh)  (per-m, 2048x64x512)
    gemm_bf16_wmma<128, 128, 1, true><<<dim3(512 / 128, 2048 / 128, 16), 256, 0, stream>>>(
        zz, wzhT, bzh, h1, 64, 512,
        2048LL * 64, 512LL * 64, 512LL, 2048LL * 512);

    // ---- stage 3: h2 = relu(h1 @ Whd + bhd)  (32768x512x512)
    gemm_bf16_wmma<128, 128, 1, true><<<dim3(512 / 128, (unsigned)(ROWS / 128), 1), 256, 0, stream>>>(
        h1, whdT, bhd, h2, 512, 512,
        0LL, 0LL, 0LL, 0LL);

    // ---- stage 4: out = sigmoid(h2 @ Whx + bhx)  (32768x512x3072, f32 out)
    gemm_bf16_wmma<128, 128, 2, false><<<dim3(3072 / 128, (unsigned)(ROWS / 128), 1), 256, 0, stream>>>(
        h2, whxT, bhx, out, 512, 3072,
        0LL, 0LL, 0LL, 0LL);
}
